// TopKSAE_71725953843965
// MI455X (gfx1250) — compile-verified, matrix operand order checked by reading, not run on hardware
//
#include <hip/hip_runtime.h>
#include <hip/hip_bf16.h>

#define D_MODEL 768
#define N_LAT   24576
#define K_ACT   32
#define BATCH   8192

typedef __bf16 bf16_t;
typedef __attribute__((ext_vector_type(16))) __bf16 v16bf;
typedef __attribute__((ext_vector_type(8)))  float  v8f;

union FragU { uint4 q[2]; v16bf v; };

// ---------------------------------------------------------------------------
// Kernel 1: xb = bf16(x - b_pre)   [BATCH, D_MODEL]
// ---------------------------------------------------------------------------
__global__ __launch_bounds__(256) void prep_xb(const float* __restrict__ x,
                                               const float* __restrict__ b_pre,
                                               bf16_t* __restrict__ xb) {
    int i = blockIdx.x * 256 + threadIdx.x;
    if (i < BATCH * D_MODEL) {
        int d = i % D_MODEL;
        xb[i] = (bf16_t)(x[i] - b_pre[d]);
    }
}

// ---------------------------------------------------------------------------
// Kernel 2: we = bf16(W_enc)       [N_LAT, D_MODEL]
// ---------------------------------------------------------------------------
__global__ __launch_bounds__(256) void prep_we(const float* __restrict__ w,
                                               bf16_t* __restrict__ wb) {
    int i = blockIdx.x * 256 + threadIdx.x;
    if (i < N_LAT * D_MODEL) {
        wb[i] = (bf16_t)(w[i]);
    }
}

// ---------------------------------------------------------------------------
// Kernel 3: WdT[l, d] = W_dec[d, l]  (LDS-tiled fp32 transpose, 32x32 tiles)
// ---------------------------------------------------------------------------
__global__ __launch_bounds__(256) void transpose_wdec(const float* __restrict__ Wd,
                                                      float* __restrict__ WdT) {
    __shared__ float tile[32][33];
    const int bx = blockIdx.x * 32;            // l dimension
    const int by = blockIdx.y * 32;            // d dimension
    const int tx = threadIdx.x & 31;
    const int ty = threadIdx.x >> 5;           // 0..7
    #pragma unroll
    for (int i = 0; i < 32; i += 8)
        tile[ty + i][tx] = Wd[(size_t)(by + ty + i) * N_LAT + (bx + tx)];
    __syncthreads();
    #pragma unroll
    for (int i = 0; i < 32; i += 8)
        WdT[(size_t)(bx + ty + i) * D_MODEL + (by + tx)] = tile[tx][ty + i];
}

// ---------------------------------------------------------------------------
// Kernel 4: H = Xb @ We^T via v_wmma_f32_16x16x32_bf16
//   Block: 256 threads = 8 waves; block tile 128(M) x 128(N).
//   Wave grid 4(M) x 2(N): each wave owns a 32x64 subtile =
//   2 A-fragments x 4 B-fragments -> 8 WMMAs per 32-wide K-step.
//   A fragment (16x32 bf16): lane m = lane&15; lanes 0-15 hold K
//   {0..7,16..23}, lanes 16-31 hold K {8..15,24..31} (two 16B loads/lane).
//   B fragment mirrors this per-lane layout from We rows.
// ---------------------------------------------------------------------------
__device__ inline v16bf load_frag16x32(const bf16_t* __restrict__ base,
                                       int row, int kBase, int lane) {
    const int m   = row + (lane & 15);
    const int kHi = (lane >> 4) << 3;  // 0 or 8
    const bf16_t* p = base + (size_t)m * D_MODEL + kBase + kHi;
    FragU f;
    f.q[0] = *(const uint4*)(p);        // K = kHi+0 .. kHi+7
    f.q[1] = *(const uint4*)(p + 16);   // K = kHi+16 .. kHi+23
    return f.v;
}

__global__ __launch_bounds__(256) void sae_encode_gemm(
        const bf16_t* __restrict__ Xb,   // [BATCH, D_MODEL]
        const bf16_t* __restrict__ We,   // [N_LAT, D_MODEL]
        float* __restrict__ H) {         // [BATCH, N_LAT]
    const int lane = threadIdx.x & 31;
    const int wave = threadIdx.x >> 5;   // 0..7
    const int wm   = wave & 3;           // 4 M-slices of 32 rows
    const int wn   = wave >> 2;          // 2 N-slices of 64 cols
    const int rowBase = blockIdx.y * 128 + wm * 32;
    const int colBase = blockIdx.x * 128 + wn * 64;

    v8f acc[2][4] = {};

    for (int k = 0; k < D_MODEL; k += 32) {
        v16bf a[2], b[4];
        #pragma unroll
        for (int i = 0; i < 2; ++i)
            a[i] = load_frag16x32(Xb, rowBase + i * 16, k, lane);
        #pragma unroll
        for (int j = 0; j < 4; ++j)
            b[j] = load_frag16x32(We, colBase + j * 16, k, lane);

        if (k + 32 < D_MODEL) {  // hint next K-step into cache
            __builtin_prefetch(Xb + (size_t)(rowBase + (lane & 15)) * D_MODEL + k + 32, 0, 1);
            __builtin_prefetch(We + (size_t)(colBase + (lane & 15)) * D_MODEL + k + 32, 0, 1);
        }

        #pragma unroll
        for (int i = 0; i < 2; ++i)
            #pragma unroll
            for (int j = 0; j < 4; ++j)
                acc[i][j] = __builtin_amdgcn_wmma_f32_16x16x32_bf16(
                    /*neg_a=*/false, a[i], /*neg_b=*/false, b[j],
                    /*c_mod=*/(short)0, acc[i][j],
                    /*reuse_a=*/false, /*reuse_b=*/false);
    }

    // C/D layout: VGPR r holds M = r (lanes 0-15) or M = 8+r (lanes 16-31),
    // N = lane & 15.
    const int cN   = lane & 15;
    const int cMhi = (lane >> 4) << 3;
    #pragma unroll
    for (int i = 0; i < 2; ++i)
        #pragma unroll
        for (int j = 0; j < 4; ++j)
            #pragma unroll
            for (int r = 0; r < 8; ++r) {
                const int row = rowBase + i * 16 + cMhi + r;
                const int col = colBase + j * 16 + cN;
                H[(size_t)row * N_LAT + col] = acc[i][j][r];
            }
}

// ---------------------------------------------------------------------------
// Kernel 5: per-row exact top-k threshold (MSB-first radix select on the
// monotonic uint mapping), sparsify z in place, sparse decode to x_hat.
// One 256-thread block per batch row; row re-reads are L2-resident.
// ---------------------------------------------------------------------------
__device__ inline unsigned f2mono(float f) {
    unsigned u = __float_as_uint(f);
    return (u & 0x80000000u) ? ~u : (u | 0x80000000u);
}

__global__ __launch_bounds__(256) void topk_decode(
        float* __restrict__ Z,            // [BATCH, N_LAT] in: h, out: sparse z
        const float* __restrict__ WdT,    // [N_LAT, D_MODEL]
        const float* __restrict__ b_pre,  // [D_MODEL]
        float* __restrict__ Xhat) {       // [BATCH, D_MODEL]
    __shared__ unsigned hist[256];
    __shared__ int   nzIdx[64];
    __shared__ float nzVal[64];
    __shared__ unsigned s_prefix;
    __shared__ int s_krem;
    __shared__ int s_cnt;

    const int row = blockIdx.x;
    const int tid = threadIdx.x;
    float* hrow = Z + (size_t)row * N_LAT;

    if (tid == 0) { s_prefix = 0u; s_krem = K_ACT; s_cnt = 0; }
    __syncthreads();

    // 4 rounds x 8 bits, MSB first: exact k-th largest threshold.
    for (int r = 0; r < 4; ++r) {
        hist[tid] = 0u;
        __syncthreads();
        const unsigned shift = 24u - 8u * r;
        const unsigned pfx   = s_prefix;
        for (int i = tid; i < N_LAT; i += 256) {
            unsigned u = f2mono(hrow[i]);
            if (r == 0 || (u >> (shift + 8u)) == pfx)
                atomicAdd(&hist[(u >> shift) & 255u], 1u);
        }
        __syncthreads();
        if (tid == 0) {
            int krem = s_krem;
            int cum = 0;
            int b = 255;
            for (; b > 0; --b) {
                cum += (int)hist[b];
                if (cum >= krem) break;
            }
            if (cum < krem) cum += (int)hist[0];  // b == 0 fallthrough
            s_krem   = krem - (cum - (int)hist[b]);
            s_prefix = (pfx << 8) | (unsigned)b;
        }
        __syncthreads();
    }
    const unsigned thr = s_prefix;

    // Sparsify in place; collect surviving (idx, val) pairs.
    for (int i = tid; i < N_LAT; i += 256) {
        const float h = hrow[i];
        const bool keep = (f2mono(h) >= thr);
        hrow[i] = keep ? h : 0.0f;
        if (keep) {
            int slot = atomicAdd(&s_cnt, 1);
            if (slot < 64) { nzIdx[slot] = i; nzVal[slot] = h; }
        }
    }
    __syncthreads();
    const int nnz = (s_cnt < 64) ? s_cnt : 64;

    // Sparse decode: x_hat[d] = b_pre[d] + sum_j val_j * WdT[idx_j, d]
    for (int d = tid; d < D_MODEL; d += 256) {
        float acc = b_pre[d];
        for (int j = 0; j < nnz; ++j)
            acc += nzVal[j] * WdT[(size_t)nzIdx[j] * D_MODEL + d];
        Xhat[(size_t)row * D_MODEL + d] = acc;
    }
}

// ---------------------------------------------------------------------------
// Launch
// ---------------------------------------------------------------------------
extern "C" void kernel_launch(void* const* d_in, const int* in_sizes, int n_in,
                              void* d_out, int out_size, void* d_ws, size_t ws_size,
                              hipStream_t stream) {
    const float* x     = (const float*)d_in[0];  // [BATCH, D_MODEL]
    const float* W_enc = (const float*)d_in[1];  // [N_LAT, D_MODEL]
    const float* W_dec = (const float*)d_in[2];  // [D_MODEL, N_LAT]
    const float* b_pre = (const float*)d_in[3];  // [D_MODEL]

    float* x_hat = (float*)d_out;                           // [BATCH, D_MODEL]
    float* z     = (float*)d_out + (size_t)BATCH * D_MODEL; // [BATCH, N_LAT]

    // Workspace layout (all offsets 256B-aligned by construction)
    char* ws = (char*)d_ws;
    bf16_t* xb  = (bf16_t*)ws;                           // 12.6 MB
    ws += (size_t)BATCH * D_MODEL * sizeof(bf16_t);
    bf16_t* we  = (bf16_t*)ws;                           // 37.7 MB
    ws += (size_t)N_LAT * D_MODEL * sizeof(bf16_t);
    float*  wdT = (float*)ws;                            // 75.5 MB
    (void)ws_size; (void)n_in; (void)in_sizes; (void)out_size;

    prep_xb<<<(BATCH * D_MODEL + 255) / 256, 256, 0, stream>>>(x, b_pre, xb);
    prep_we<<<(N_LAT * D_MODEL + 255) / 256, 256, 0, stream>>>(W_enc, we);
    transpose_wdec<<<dim3(N_LAT / 32, D_MODEL / 32), 256, 0, stream>>>(W_dec, wdT);
    sae_encode_gemm<<<dim3(N_LAT / 128, BATCH / 128), 256, 0, stream>>>(xb, we, z);
    topk_decode<<<BATCH, 256, 0, stream>>>(z, wdT, b_pre, x_hat);
}
